// CapgMyoNet_50620484551185
// MI455X (gfx1250) — matrix operand order repaired and checked by reading, not compiled, not run on hardware
//
#include <hip/hip_runtime.h>

// ---------------------------------------------------------------------------
// CapgMyoNet fused inference for MI455X (gfx1250), wave32 + WMMA bf16.
//   Heavy stages (conv2, lc3, lc4, fc5..fc8) are bf16 WMMA GEMMs with f32
//   accumulation (v_wmma_f32_16x16x32_bf16), register-blocked 4x in N.
//   All 4 B-fragments are loaded into distinct registers BEFORE the WMMA
//   chain so the 4 independent WMMAs issue back-to-back while the next
//   loads are in flight (avoids s_wait_loadcnt 0 before every WMMA).
//   FC stages stage the activation tile through LDS (global_load_b128 ->
//   ds_load_b128 shared by all 4 waves).
//   Activation layout: [b][pixel][c] (c contiguous) -> all fragment loads and
//   epilogue stores are aligned 16-byte b128 ops.
// ---------------------------------------------------------------------------

typedef __bf16 v16bf __attribute__((ext_vector_type(16)));
typedef float  v8f   __attribute__((ext_vector_type(8)));

#define HHW 8
#define WWW 16
#define NPIX 128          // H*W
#define CCH 64            // channels
#define FEAT 8192         // C*H*W
#define KBLK 256          // FC LDS K-chunk
#define NROWS 64          // FC batch rows staged in LDS

union FragBF { uint4 q[2]; v16bf v; };

// A-fragment (16x32 bf16), W row-major [M][ldk].
// Lane<16 holds K kb..kb+15; lane>=16 holds kb+16..kb+31.
__device__ __forceinline__ v16bf load_a_frag(const __bf16* __restrict__ W,
                                             int ldk, int mbase, int kbase) {
  const int l   = threadIdx.x & 31;
  const int row = mbase + (l & 15);
  const int k0  = kbase + ((l & 16) ? 16 : 0);
  const __bf16* base = W + (size_t)row * (size_t)ldk;
  FragBF f;
  f.q[0] = *(const uint4*)(base + k0);        // K = k0 .. k0+7
  f.q[1] = *(const uint4*)(base + k0 + 8);    // K = k0+8 .. k0+15
  return f.v;
}

// B-fragment (32x16 bf16) from global; act stored K-contiguous per column.
// Lane<16 holds K kb..kb+15; lane>=16 holds kb+16..kb+31.
__device__ __forceinline__ v16bf load_b_frag(const __bf16* __restrict__ A,
                                             size_t ldk, size_t nbase,
                                             size_t koff, int kbase) {
  const int l   = threadIdx.x & 31;
  const size_t col = nbase + (size_t)(l & 15);
  const int k0  = kbase + ((l & 16) ? 16 : 0);
  const __bf16* base = A + col * ldk + koff;
  FragBF f;
  f.q[0] = *(const uint4*)(base + k0);
  f.q[1] = *(const uint4*)(base + k0 + 8);
  return f.v;
}

// B-fragment from LDS tile [rows][KBLK].
__device__ __forceinline__ v16bf load_b_frag_lds(const __bf16* lds, int rowbase,
                                                 int kbase) {
  const int l   = threadIdx.x & 31;
  const int row = rowbase + (l & 15);
  const int k0  = kbase + ((l & 16) ? 16 : 0);
  const __bf16* base = lds + row * KBLK + k0;
  FragBF f;
  f.q[0] = *(const uint4*)(base);
  f.q[1] = *(const uint4*)(base + 8);
  return f.v;
}

__device__ __forceinline__ v8f wmma_bf16(v16bf a, v16bf b, v8f c) {
  return __builtin_amdgcn_wmma_f32_16x16x32_bf16(false, a, false, b,
                                                 (short)0, c, false, false);
}

__device__ __forceinline__ v8f vzero8() {
  v8f z = {0.f, 0.f, 0.f, 0.f, 0.f, 0.f, 0.f, 0.f};
  return z;
}

// ---------------------------------------------------------------------------
// Prep: fold BN params into scale/bias.
// prm layout (floats): s1[0..63] t1[64..127] s2[128] t2[192] s3[256] t3[320]
//                      s4[384] t4[448] | 512:s0 513:t0 514:s5 515:t5
//                      516:s6 517:t6 518:s7 519:t7
// ---------------------------------------------------------------------------
__global__ void prep_params_kernel(const float* bn0, const float* bn1,
                                   const float* bn2, const float* bn3,
                                   const float* bn4, const float* bn5,
                                   const float* bn6, const float* bn7,
                                   float* prm) {
  const int c = threadIdx.x;
  if (c < CCH) {
    const float* bns[4] = {bn1, bn2, bn3, bn4};
    for (int i = 0; i < 4; ++i) {
      const float* p = bns[i];
      float g = p[c], be = p[CCH + c], m = p[2 * CCH + c], v = p[3 * CCH + c];
      float s = g * rsqrtf(v + 1e-5f);
      prm[i * 128 + c]      = s;
      prm[i * 128 + 64 + c] = be - m * s;
    }
  }
  if (c == 0) {
    const float* sc[4] = {bn0, bn5, bn6, bn7};
    for (int i = 0; i < 4; ++i) {
      float g = sc[i][0], be = sc[i][1], m = sc[i][2], v = sc[i][3];
      float s = g * rsqrtf(v + 1e-5f);
      prm[512 + 2 * i] = s;
      prm[513 + 2 * i] = be - m * s;
    }
  }
}

// conv2_w (o,c,ky,kx) f32 -> [tap][o][c] bf16
__global__ void perm_conv2_kernel(const float* __restrict__ w,
                                  __bf16* __restrict__ o) {
  int idx = blockIdx.x * 256 + threadIdx.x;
  if (idx >= CCH * CCH * 9) return;
  int tap = idx % 9, oc = idx / 9;
  o[tap * 4096 + oc] = (__bf16)w[idx];
}

// lc_w (o,c,h,w) f32 -> [pixel][o][c] bf16
__global__ void perm_lc_kernel(const float* __restrict__ w,
                               __bf16* __restrict__ o) {
  int idx = blockIdx.x * 256 + threadIdx.x;
  if (idx >= CCH * CCH * NPIX) return;
  int p = idx & 127, oc = idx >> 7;
  o[p * 4096 + oc] = (__bf16)w[idx];
}

// fc5_w (512, f=c*128+p) f32 -> [j][g=p*64+c] bf16 (K reordered to act layout)
__global__ void perm_fc5_kernel(const float* __restrict__ w,
                                __bf16* __restrict__ o) {
  int idx = blockIdx.x * 256 + threadIdx.x;
  if (idx >= 512 * FEAT) return;
  int g = idx & (FEAT - 1), j = idx >> 13;
  int c = g & 63, p = g >> 6;
  o[idx] = (__bf16)w[(size_t)j * FEAT + c * NPIX + p];
}

__global__ void cvt_kernel(const float* __restrict__ w, __bf16* __restrict__ o,
                           int n) {
  int idx = blockIdx.x * 256 + threadIdx.x;
  if (idx < n) o[idx] = (__bf16)w[idx];
}

// fc8_w (8,128) -> zero-padded (16,128) bf16 so WMMA M-tile is full
__global__ void pad_fc8_kernel(const float* __restrict__ w,
                               __bf16* __restrict__ o) {
  int idx = blockIdx.x * 256 + threadIdx.x;
  if (idx >= 16 * 128) return;
  int j = idx >> 7;
  o[idx] = (j < 8) ? (__bf16)w[idx] : (__bf16)0.f;
}

// ---------------------------------------------------------------------------
// Stage 1: permute input + bn0 + conv1 (1->64) + bn1 + relu -> bf16 act
// h(y,x) = x_flat[b*128 + x*8 + y]  (view(B,16,8).transpose)
// ---------------------------------------------------------------------------
__global__ void conv1_kernel(const float* __restrict__ xin,
                             const float* __restrict__ w1,   // (64,1,3,3)
                             const float* __restrict__ b1,
                             const float* __restrict__ prm,
                             __bf16* __restrict__ actOut, int Btot) {
  int tid = blockIdx.x * blockDim.x + threadIdx.x;
  if (tid >= Btot * NPIX) return;
  const int p = tid & 127;
  const size_t b = (size_t)(tid >> 7);
  const int y = p >> 4, x = p & 15;
  const float s0 = prm[512], t0 = prm[513];
  const float* s1 = prm;
  const float* t1 = prm + 64;

  float in[9];
#pragma unroll
  for (int dy = -1; dy <= 1; ++dy)
#pragma unroll
    for (int dx = -1; dx <= 1; ++dx) {
      int yy = y + dy, xx = x + dx;
      float v = 0.f;
      if (yy >= 0 && yy < HHW && xx >= 0 && xx < WWW)
        v = s0 * xin[b * NPIX + xx * HHW + yy] + t0;
      in[(dy + 1) * 3 + (dx + 1)] = v;
    }

  for (int og = 0; og < CCH; og += 8) {
    union { uint4 qv; __bf16 h[8]; } u;
#pragma unroll
    for (int i = 0; i < 8; ++i) {
      int o = og + i;
      float acc = b1[o];
#pragma unroll
      for (int tap = 0; tap < 9; ++tap) acc += w1[o * 9 + tap] * in[tap];
      u.h[i] = (__bf16)fmaxf(acc * s1[o] + t1[o], 0.f);
    }
    *(uint4*)(actOut + b * FEAT + (size_t)p * CCH + og) = u.qv;
  }
}

// ---------------------------------------------------------------------------
// Stage 2: conv2 (64->64, 3x3 SAME) as per-pixel 9-tap WMMA GEMM + bn2 + relu
// block = 128 thr (4 waves = 4 M-tiles of 16 out-ch), grid = (128 pixels, B/64)
// Per (tap, k-step): 1 A-fragment + 4 B-fragments loaded up-front, then 4
// independent back-to-back WMMAs.
// ---------------------------------------------------------------------------
__global__ void conv2_wmma_kernel(const __bf16* __restrict__ actIn,
                                  const __bf16* __restrict__ w2p, // [9][64][64]
                                  const float* __restrict__ cb,   // conv2_b
                                  const float* __restrict__ prm,
                                  __bf16* __restrict__ actOut) {
  const int pixel = blockIdx.x;
  const int y = pixel >> 4, x = pixel & 15;
  const int mbase = (threadIdx.x >> 5) * 16;
  const size_t batchBase = (size_t)blockIdx.y * 64;
  const float* s = prm + 128;
  const float* t = prm + 192;
  const int l = threadIdx.x & 31;
  const int chan = mbase + ((l & 16) ? 8 : 0);

  v8f acc[4] = {vzero8(), vzero8(), vzero8(), vzero8()};

  for (int dy = -1; dy <= 1; ++dy) {
    const int yy = y + dy;
    if (yy < 0 || yy >= HHW) continue;
    for (int dx = -1; dx <= 1; ++dx) {
      const int xx = x + dx;
      if (xx < 0 || xx >= WWW) continue;
      const int tap = (dy + 1) * 3 + (dx + 1);
      const int q = yy * WWW + xx;
      const __bf16* wt = w2p + tap * 4096;
#pragma unroll
      for (int kb = 0; kb < CCH; kb += 32) {
        v16bf a = load_a_frag(wt, CCH, mbase, kb);
        v16bf bf[4];
#pragma unroll
        for (int nb = 0; nb < 4; ++nb)
          bf[nb] = load_b_frag(actIn, FEAT, batchBase + nb * 16,
                               (size_t)q * CCH, kb);
#pragma unroll
        for (int nb = 0; nb < 4; ++nb)
          acc[nb] = wmma_bf16(a, bf[nb], acc[nb]);
      }
    }
  }
#pragma unroll
  for (int nb = 0; nb < 4; ++nb) {
    const size_t bidx = batchBase + nb * 16 + (size_t)(l & 15);
    union { uint4 qv; __bf16 h[8]; } u;
#pragma unroll
    for (int i = 0; i < 8; ++i) {
      float v = acc[nb][i] + cb[chan + i];
      u.h[i] = (__bf16)fmaxf(v * s[chan + i] + t[chan + i], 0.f);
    }
    *(uint4*)(actOut + bidx * FEAT + (size_t)pixel * CCH + chan) = u.qv;
  }
}

// ---------------------------------------------------------------------------
// Stage 3/4: locally-connected k=1 (per-pixel 64x64) WMMA GEMM + bn + relu
// ---------------------------------------------------------------------------
__global__ void lc_wmma_kernel(const __bf16* __restrict__ actIn,
                               const __bf16* __restrict__ wp, // [128][64][64]
                               const float* __restrict__ s,
                               const float* __restrict__ t,
                               __bf16* __restrict__ actOut) {
  const int pixel = blockIdx.x;
  const int mbase = (threadIdx.x >> 5) * 16;
  const size_t batchBase = (size_t)blockIdx.y * 64;
  const __bf16* wt = wp + (size_t)pixel * 4096;
  const int l = threadIdx.x & 31;
  const int chan = mbase + ((l & 16) ? 8 : 0);

  v8f acc[4] = {vzero8(), vzero8(), vzero8(), vzero8()};
#pragma unroll
  for (int kb = 0; kb < CCH; kb += 32) {
    v16bf a = load_a_frag(wt, CCH, mbase, kb);
    v16bf bf[4];
#pragma unroll
    for (int nb = 0; nb < 4; ++nb)
      bf[nb] = load_b_frag(actIn, FEAT, batchBase + nb * 16,
                           (size_t)pixel * CCH, kb);
#pragma unroll
    for (int nb = 0; nb < 4; ++nb)
      acc[nb] = wmma_bf16(a, bf[nb], acc[nb]);
  }
#pragma unroll
  for (int nb = 0; nb < 4; ++nb) {
    const size_t bidx = batchBase + nb * 16 + (size_t)(l & 15);
    union { uint4 qv; __bf16 h[8]; } u;
#pragma unroll
    for (int i = 0; i < 8; ++i)
      u.h[i] = (__bf16)fmaxf(acc[nb][i] * s[chan + i] + t[chan + i], 0.f);
    *(uint4*)(actOut + bidx * FEAT + (size_t)pixel * CCH + chan) = u.qv;
  }
}

// ---------------------------------------------------------------------------
// FC: out[b][j] = relu(s*(W[j,:]·act[b,:] + bias[j]) + t), bf16 out.
// block 128 (4 waves = 64 out rows), grid (J/64, B/64).
// Activation tile (64 batches x KBLK) staged in LDS; B-fragments come from
// ds_load_b128, batched 4-wide ahead of the WMMA chain.
// ---------------------------------------------------------------------------
__global__ void fc_bn_relu_kernel(const __bf16* __restrict__ actIn, // [B][K]
                                  const __bf16* __restrict__ W,     // [J][K]
                                  const float* __restrict__ bias,   // [J]
                                  const float* __restrict__ snt,    // {s,t}
                                  __bf16* __restrict__ actOut,      // [B][J]
                                  int K, int J) {
  __shared__ __bf16 smem[NROWS * KBLK];   // 32 KB
  const int mbase = blockIdx.x * 64 + (threadIdx.x >> 5) * 16;
  const size_t nbase = (size_t)blockIdx.y * NROWS;
  const int tid = threadIdx.x;

  v8f acc[4] = {vzero8(), vzero8(), vzero8(), vzero8()};

  for (int k0 = 0; k0 < K; k0 += KBLK) {
    __syncthreads();
    // cooperative stage: NROWS x KBLK bf16 = 2048 uint4, 128 threads x 16
#pragma unroll
    for (int i = 0; i < (NROWS * KBLK / 8) / 128; ++i) {
      int idx = i * 128 + tid;
      int row = idx >> 5;          // KBLK/8 = 32 uint4 per row
      int c16 = idx & 31;
      ((uint4*)smem)[idx] =
          *(const uint4*)(actIn + (nbase + row) * (size_t)K + k0 + c16 * 8);
    }
    __syncthreads();
#pragma unroll
    for (int kb = 0; kb < KBLK; kb += 32) {
      v16bf a = load_a_frag(W, K, mbase, k0 + kb);
      v16bf bf[4];
#pragma unroll
      for (int nb = 0; nb < 4; ++nb)
        bf[nb] = load_b_frag_lds(smem, nb * 16, kb);
#pragma unroll
      for (int nb = 0; nb < 4; ++nb)
        acc[nb] = wmma_bf16(a, bf[nb], acc[nb]);
    }
  }

  const float s = snt[0], tt = snt[1];
  const int l = threadIdx.x & 31;
  const int chan = mbase + ((l & 16) ? 8 : 0);
#pragma unroll
  for (int nb = 0; nb < 4; ++nb) {
    const size_t bidx = nbase + nb * 16 + (size_t)(l & 15);
    union { uint4 qv; __bf16 h[8]; } u;
#pragma unroll
    for (int i = 0; i < 8; ++i) {
      float v = acc[nb][i] + bias[chan + i];
      u.h[i] = (__bf16)fmaxf(v * s + tt, 0.f);
    }
    *(uint4*)(actOut + bidx * (size_t)J + chan) = u.qv;
  }
}

// fc8: (8,128) GEMM (M padded to 16), f32 output (B,8). One wave per WG.
__global__ void fc_final_kernel(const __bf16* __restrict__ actIn, // [B][128]
                                const __bf16* __restrict__ W,     // [16][128]
                                const float* __restrict__ bias,   // [8]
                                float* __restrict__ out) {
  const size_t nbase = (size_t)blockIdx.x * 16;
  v8f acc = vzero8();
#pragma unroll
  for (int kb = 0; kb < 128; kb += 32) {
    v16bf a = load_a_frag(W, 128, 0, kb);
    v16bf b = load_b_frag(actIn, 128, nbase, 0, kb);
    acc = wmma_bf16(a, b, acc);
  }
  const int l = threadIdx.x & 31;
  if (l < 16) {  // rows 0..7 == valid classes; lanes>=16 hold rows 8..15 (pad)
    const size_t bidx = nbase + (size_t)l;
#pragma unroll
    for (int i = 0; i < 8; ++i) out[bidx * 8 + i] = acc[i] + bias[i];
  }
}

// ---------------------------------------------------------------------------
extern "C" void kernel_launch(void* const* d_in, const int* in_sizes, int n_in,
                              void* d_out, int out_size, void* d_ws,
                              size_t ws_size, hipStream_t stream) {
  const float* x      = (const float*)d_in[0];
  const float* bn0    = (const float*)d_in[1];
  const float* conv1w = (const float*)d_in[2];
  const float* conv1b = (const float*)d_in[3];
  const float* bn1    = (const float*)d_in[4];
  const float* conv2w = (const float*)d_in[5];
  const float* conv2b = (const float*)d_in[6];
  const float* bn2    = (const float*)d_in[7];
  const float* lc3w   = (const float*)d_in[8];
  const float* bn3    = (const float*)d_in[9];
  const float* lc4w   = (const float*)d_in[10];
  const float* bn4    = (const float*)d_in[11];
  const float* fc5w   = (const float*)d_in[12];
  const float* fc5b   = (const float*)d_in[13];
  const float* bn5    = (const float*)d_in[14];
  const float* fc6w   = (const float*)d_in[15];
  const float* fc6b   = (const float*)d_in[16];
  const float* bn6    = (const float*)d_in[17];
  const float* fc7w   = (const float*)d_in[18];
  const float* fc7b   = (const float*)d_in[19];
  const float* bn7    = (const float*)d_in[20];
  const float* fc8w   = (const float*)d_in[21];
  const float* fc8b   = (const float*)d_in[22];

  const int B = in_sizes[0] / NPIX;  // 8192 in reference

  char* ws = (char*)d_ws;
  size_t off = 0;
  auto carve = [&](size_t bytes) -> char* {
    char* p = ws + off;
    off = (off + bytes + 255) & ~(size_t)255;
    return p;
  };
  float*  prm  = (float*)carve(1024 * sizeof(float));
  __bf16* w2p  = (__bf16*)carve((size_t)9 * 4096 * 2);
  __bf16* lc3p = (__bf16*)carve((size_t)NPIX * 4096 * 2);
  __bf16* lc4p = (__bf16*)carve((size_t)NPIX * 4096 * 2);
  __bf16* w5p  = (__bf16*)carve((size_t)512 * FEAT * 2);
  __bf16* w6p  = (__bf16*)carve((size_t)512 * 512 * 2);
  __bf16* w7p  = (__bf16*)carve((size_t)128 * 512 * 2);
  __bf16* w8p  = (__bf16*)carve((size_t)16 * 128 * 2);
  __bf16* bufA = (__bf16*)carve((size_t)B * FEAT * 2);
  __bf16* bufB = (__bf16*)carve((size_t)B * FEAT * 2);
  __bf16* act5 = (__bf16*)carve((size_t)B * 512 * 2);
  __bf16* act6 = (__bf16*)carve((size_t)B * 512 * 2);
  __bf16* act7 = (__bf16*)carve((size_t)B * 128 * 2);

  // ---- prep ----
  prep_params_kernel<<<1, 64, 0, stream>>>(bn0, bn1, bn2, bn3, bn4, bn5, bn6,
                                           bn7, prm);
  perm_conv2_kernel<<<(9 * 4096 + 255) / 256, 256, 0, stream>>>(conv2w, w2p);
  perm_lc_kernel<<<(NPIX * 4096 + 255) / 256, 256, 0, stream>>>(lc3w, lc3p);
  perm_lc_kernel<<<(NPIX * 4096 + 255) / 256, 256, 0, stream>>>(lc4w, lc4p);
  perm_fc5_kernel<<<(512 * FEAT + 255) / 256, 256, 0, stream>>>(fc5w, w5p);
  cvt_kernel<<<(512 * 512 + 255) / 256, 256, 0, stream>>>(fc6w, w6p, 512 * 512);
  cvt_kernel<<<(128 * 512 + 255) / 256, 256, 0, stream>>>(fc7w, w7p, 128 * 512);
  pad_fc8_kernel<<<(16 * 128 + 255) / 256, 256, 0, stream>>>(fc8w, w8p);

  // ---- network ----
  conv1_kernel<<<(B * NPIX + 255) / 256, 256, 0, stream>>>(x, conv1w, conv1b,
                                                           prm, bufA, B);
  conv2_wmma_kernel<<<dim3(NPIX, B / 64), 128, 0, stream>>>(bufA, w2p, conv2b,
                                                            prm, bufB);
  lc_wmma_kernel<<<dim3(NPIX, B / 64), 128, 0, stream>>>(bufB, lc3p, prm + 256,
                                                         prm + 320, bufA);
  lc_wmma_kernel<<<dim3(NPIX, B / 64), 128, 0, stream>>>(bufA, lc4p, prm + 384,
                                                         prm + 448, bufB);
  fc_bn_relu_kernel<<<dim3(512 / 64, B / NROWS), 128, 0, stream>>>(
      bufB, w5p, fc5b, prm + 514, act5, FEAT, 512);
  fc_bn_relu_kernel<<<dim3(512 / 64, B / NROWS), 128, 0, stream>>>(
      act5, w6p, fc6b, prm + 516, act6, 512, 512);
  fc_bn_relu_kernel<<<dim3(128 / 64, B / NROWS), 128, 0, stream>>>(
      act6, w7p, fc7b, prm + 518, act7, 512, 128);
  fc_final_kernel<<<B / 16, 32, 0, stream>>>(act7, w8p, fc8b, (float*)d_out);
}
